// DPINet_70746701300333
// MI455X (gfx1250) — compile-verified, matrix operand order checked by reading, not run on hardware
//
#include <hip/hip_runtime.h>
#include <hip/hip_bf16.h>

#define NPART 65536
#define NEDGE 524288
#define RIGC  4096
#define NF    128
#define DT_INV 60.0f

typedef __attribute__((ext_vector_type(16))) __bf16 v16bf;
typedef __attribute__((ext_vector_type(8)))  __bf16 v8bf;
typedef __attribute__((ext_vector_type(8)))  float  v8f;

// CDNA5 16-bit A/B fragment K-offset for element e (0..15),
// lane half hi (0 = lanes 0-15, 1 = lanes 16-31):
//   e in 0..7  -> k = kbase + 8*hi + e          (contiguous 16 bytes)
//   e in 8..15 -> k = kbase + 16 + 8*hi + (e-8) (contiguous 16 bytes)
// => one fragment from a bf16 row-major buffer is exactly two b128 loads.
__device__ __forceinline__ int koff(int e, int hi) {
  int v = e >> 1, j = e & 1;
  return ((v & 4) ? 16 : 0) + hi * 8 + (v & 3) * 2 + j;
}

__device__ __forceinline__ v8f vzero8() {
  v8f z = {0.f, 0.f, 0.f, 0.f, 0.f, 0.f, 0.f, 0.f};
  return z;
}

__device__ __forceinline__ v8f wmma_bf16(v16bf a, v16bf b, v8f c) {
  return __builtin_amdgcn_wmma_f32_16x16x32_bf16(false, a, false, b, (short)0, c,
                                                 false, false);
}

// Fragment load from bf16 row-major storage: `row` points at the row start,
// element k = kbase + koff(e, hi).  Two 16-byte loads (global or LDS).
__device__ __forceinline__ v16bf frag_ld(const __bf16* row, int kbase, int hi) {
  const v8bf* p0 = (const v8bf*)(row + kbase + 8 * hi);
  const v8bf* p1 = (const v8bf*)(row + kbase + 8 * hi + 16);
  v8bf lo = *p0;
  v8bf hi8 = *p1;
  return __builtin_shufflevector(lo, hi8, 0, 1, 2, 3, 4, 5, 6, 7, 8, 9, 10, 11, 12,
                                 13, 14, 15);
}

// ---------------------------------------------------------------------------
__global__ void zero_f32_kernel(float* __restrict__ p, long n) {
  long i = (long)blockIdx.x * blockDim.x + threadIdx.x;
  long stride = (long)gridDim.x * blockDim.x;
  for (; i < n; i += stride) p[i] = 0.0f;
}

__global__ void zero_bf16_kernel(__bf16* __restrict__ p, long n) {
  long i = (long)blockIdx.x * blockDim.x + threadIdx.x;
  long stride = (long)gridDim.x * blockDim.x;
  for (; i < n; i += stride) p[i] = (__bf16)0.0f;
}

__global__ void cvt_f32_bf16_kernel(const float* __restrict__ src,
                                    __bf16* __restrict__ dst, long n) {
  long i = (long)blockIdx.x * blockDim.x + threadIdx.x;
  long stride = (long)gridDim.x * blockDim.x;
  for (; i < n; i += stride) dst[i] = (__bf16)src[i];
}

// Convert torch-Linear W[Out][Kin] (f32) into zero-padded bf16 [Outpad][Kpad].
__global__ void convert_w_kernel(const float* __restrict__ W, __bf16* __restrict__ dst,
                                 int Out, int Kin, int Outpad, int Kpad) {
  long total = (long)Outpad * Kpad;
  long i = (long)blockIdx.x * blockDim.x + threadIdx.x;
  long stride = (long)gridDim.x * blockDim.x;
  for (; i < total; i += stride) {
    int r = (int)(i / Kpad), k = (int)(i % Kpad);
    float x = (r < Out && k < Kin) ? W[(long)r * Kin + k] : 0.0f;
    dst[i] = (__bf16)x;
  }
}

// Mean of state[:RIG] over all 6 dims -> cent[0..5]
__global__ __launch_bounds__(256) void centroid_kernel(const float* __restrict__ state,
                                                       float* __restrict__ cent) {
  __shared__ float red[256 * 6];
  int t = threadIdx.x;
  float s[6] = {0.f, 0.f, 0.f, 0.f, 0.f, 0.f};
  for (int r = t; r < RIGC; r += 256) {
#pragma unroll
    for (int d = 0; d < 6; ++d) s[d] += state[r * 6 + d];
  }
#pragma unroll
  for (int d = 0; d < 6; ++d) red[t * 6 + d] = s[d];
  __syncthreads();
  for (int off = 128; off > 0; off >>= 1) {
    if (t < off) {
#pragma unroll
      for (int d = 0; d < 6; ++d) red[t * 6 + d] += red[(t + off) * 6 + d];
    }
    __syncthreads();
  }
  if (t < 6) cent[t] = red[t] * (1.0f / RIGC);
}

// Particle encoder: pe = relu(L1(relu(L0([attr, offset, state]))))  (K0 = 15)
__global__ __launch_bounds__(128) void particle_encode_kernel(
    const float* __restrict__ state, const float* __restrict__ attr,
    const float* __restrict__ cent, const __bf16* __restrict__ w0,
    const float* __restrict__ b0, const __bf16* __restrict__ w1,
    const float* __restrict__ b1, __bf16* __restrict__ pe) {
  __shared__ __bf16 lds[4][16 * NF];
  int wave = threadIdx.x >> 5, lane = threadIdx.x & 31;
  int m = lane & 15, hi = (lane >> 4) & 1, n = m;
  __bf16* t = lds[wave];
  long rowbase = ((long)blockIdx.x * 4 + wave) * 16;

  // Layer 0: A gathered on the fly (K = 15, padded to 32)
  v16bf a;
  {
    long p = rowbase + m;
    bool rig = p < RIGC;
#pragma unroll
    for (int e = 0; e < 16; ++e) {
      int k = koff(e, hi);
      float x = 0.0f;
      if (k < 3)       x = attr[p * 3 + k];
      else if (k < 9)  x = rig ? state[p * 6 + (k - 3)] - cent[k - 3] : 0.0f;
      else if (k < 15) x = state[p * 6 + (k - 9)];
      a[e] = (__bf16)x;
    }
  }
#pragma unroll
  for (int nt = 0; nt < 8; ++nt) {
    v8f c = vzero8();
    c = wmma_bf16(a, frag_ld(w0 + (nt * 16 + n) * 32, 0, hi), c);
    float bb = b0[nt * 16 + n];
#pragma unroll
    for (int r = 0; r < 8; ++r)
      t[(r + 8 * hi) * NF + nt * 16 + n] = (__bf16)fmaxf(c[r] + bb, 0.0f);
  }
  __syncthreads();

  // Layer 1: K = 128 from LDS tile
  v8f acc[8];
#pragma unroll
  for (int nt = 0; nt < 8; ++nt) acc[nt] = vzero8();
  for (int kt = 0; kt < 4; ++kt) {
    v16bf af = frag_ld(t + m * NF, kt * 32, hi);
#pragma unroll
    for (int nt = 0; nt < 8; ++nt)
      acc[nt] = wmma_bf16(af, frag_ld(w1 + (nt * 16 + n) * NF, kt * 32, hi), acc[nt]);
  }
#pragma unroll
  for (int nt = 0; nt < 8; ++nt) {
    float bb = b1[nt * 16 + n];
#pragma unroll
    for (int r = 0; r < 8; ++r)
      pe[(rowbase + r + 8 * hi) * NF + nt * 16 + n] = (__bf16)fmaxf(acc[nt][r] + bb, 0.0f);
  }
}

// Relation encoder: rel_enc = relu(L2(relu(L1(relu(L0(rel_in))))))  (K0 = 31)
__global__ __launch_bounds__(128) void rel_encode_kernel(
    const float* __restrict__ attr, const float* __restrict__ state,
    const float* __restrict__ Ra, const int* __restrict__ recv,
    const int* __restrict__ send, const float* __restrict__ cent,
    const __bf16* __restrict__ w0, const float* __restrict__ b0,
    const __bf16* __restrict__ w1, const float* __restrict__ b1,
    const __bf16* __restrict__ w2, const float* __restrict__ b2,
    __bf16* __restrict__ rel_enc) {
  __shared__ __bf16 lds[4][16 * NF];
  int wave = threadIdx.x >> 5, lane = threadIdx.x & 31;
  int m = lane & 15, hi = (lane >> 4) & 1, n = m;
  __bf16* t = lds[wave];
  long ebase = ((long)blockIdx.x * 4 + wave) * 16;
  long e = ebase + m;
  int rv = recv[e], sd = send[e];

  // Layer 0: rel_in = [attr2[rv](9), attr2[sd](9), state[rv](6), state[sd](6), Ra]
  v16bf a;
#pragma unroll
  for (int q = 0; q < 16; ++q) {
    int k = koff(q, hi);
    float x;
    if (k < 3)        x = attr[rv * 3 + k];
    else if (k < 9)   x = (rv < RIGC) ? state[rv * 6 + (k - 3)] - cent[k - 3] : 0.0f;
    else if (k < 12)  x = attr[sd * 3 + (k - 9)];
    else if (k < 18)  x = (sd < RIGC) ? state[sd * 6 + (k - 12)] - cent[k - 12] : 0.0f;
    else if (k < 24)  x = state[rv * 6 + (k - 18)];
    else if (k < 30)  x = state[sd * 6 + (k - 24)];
    else if (k == 30) x = Ra[e];
    else              x = 0.0f;
    a[q] = (__bf16)x;
  }
#pragma unroll
  for (int nt = 0; nt < 8; ++nt) {
    v8f c = vzero8();
    c = wmma_bf16(a, frag_ld(w0 + (nt * 16 + n) * 32, 0, hi), c);
    float bb = b0[nt * 16 + n];
#pragma unroll
    for (int r = 0; r < 8; ++r)
      t[(r + 8 * hi) * NF + nt * 16 + n] = (__bf16)fmaxf(c[r] + bb, 0.0f);
  }
  __syncthreads();

  // Layers 1 and 2 (K = 128 each); layer 2 writes to global.
  for (int layer = 0; layer < 2; ++layer) {
    const __bf16* w = layer ? w2 : w1;
    const float* bv = layer ? b2 : b1;
    v8f acc[8];
#pragma unroll
    for (int nt = 0; nt < 8; ++nt) acc[nt] = vzero8();
    for (int kt = 0; kt < 4; ++kt) {
      v16bf af = frag_ld(t + m * NF, kt * 32, hi);
#pragma unroll
      for (int nt = 0; nt < 8; ++nt)
        acc[nt] = wmma_bf16(af, frag_ld(w + (nt * 16 + n) * NF, kt * 32, hi), acc[nt]);
    }
    __syncthreads();  // all reads of t complete before overwrite
#pragma unroll
    for (int nt = 0; nt < 8; ++nt) {
      float bb = bv[nt * 16 + n];
#pragma unroll
      for (int r = 0; r < 8; ++r) {
        __bf16 v = (__bf16)fmaxf(acc[nt][r] + bb, 0.0f);
        if (layer == 0)
          t[(r + 8 * hi) * NF + nt * 16 + n] = v;
        else
          rel_enc[(ebase + r + 8 * hi) * NF + nt * 16 + n] = v;
      }
    }
    __syncthreads();
  }
}

// Relation propagation: rel_eff = relu(rp([rel_enc, effect[rv], effect[sd]])),
// scatter-added into agg[rv].  K = 384.  Each wave handles TWO 16-edge tiles so
// every B fragment (rp_w) load is shared by two WMMAs (~1.25 vmem per wmma).
__global__ __launch_bounds__(128) void rel_prop_kernel(
    const __bf16* __restrict__ rel_enc, const __bf16* __restrict__ effect,
    const int* __restrict__ recv, const int* __restrict__ send,
    const __bf16* __restrict__ rp_w, const float* __restrict__ rp_b,
    float* __restrict__ agg) {
  int wave = threadIdx.x >> 5, lane = threadIdx.x & 31;
  int m = lane & 15, hi = (lane >> 4) & 1, n = m;
  long ebase = ((long)blockIdx.x * 4 + wave) * 32;
  long e0 = ebase + m, e1 = ebase + 16 + m;
  int rv0 = recv[e0], sd0 = send[e0];
  int rv1 = recv[e1], sd1 = send[e1];

  v8f acc0[8], acc1[8];
#pragma unroll
  for (int nt = 0; nt < 8; ++nt) {
    acc0[nt] = vzero8();
    acc1[nt] = vzero8();
  }
  for (int kt = 0; kt < 12; ++kt) {
    int kb = kt * 32;
    const __bf16 *arow0, *arow1;
    if (kb < NF) {
      arow0 = rel_enc + e0 * NF + kb;
      arow1 = rel_enc + e1 * NF + kb;
    } else if (kb < 2 * NF) {
      arow0 = effect + (long)rv0 * NF + (kb - NF);
      arow1 = effect + (long)rv1 * NF + (kb - NF);
    } else {
      arow0 = effect + (long)sd0 * NF + (kb - 2 * NF);
      arow1 = effect + (long)sd1 * NF + (kb - 2 * NF);
    }
    v16bf a0 = frag_ld(arow0, 0, hi);
    v16bf a1 = frag_ld(arow1, 0, hi);
#pragma unroll
    for (int nt = 0; nt < 8; ++nt) {
      v16bf b = frag_ld(rp_w + (nt * 16 + n) * (3 * NF), kb, hi);
      acc0[nt] = wmma_bf16(a0, b, acc0[nt]);
      acc1[nt] = wmma_bf16(a1, b, acc1[nt]);
    }
  }

  int tgt0[8], tgt1[8];
#pragma unroll
  for (int r = 0; r < 8; ++r) {
    tgt0[r] = recv[ebase + r + 8 * hi];
    tgt1[r] = recv[ebase + 16 + r + 8 * hi];
  }
#pragma unroll
  for (int nt = 0; nt < 8; ++nt) {
    float bb = rp_b[nt * 16 + n];
#pragma unroll
    for (int r = 0; r < 8; ++r) {
      float v0 = fmaxf(acc0[nt][r] + bb, 0.0f);
      float v1 = fmaxf(acc1[nt][r] + bb, 0.0f);
      unsafeAtomicAdd(&agg[(long)tgt0[r] * NF + nt * 16 + n], v0);
      unsafeAtomicAdd(&agg[(long)tgt1[r] * NF + nt * 16 + n], v1);
    }
  }
}

// Particle propagation: effect = relu(pp([pe, agg]))  K = 256
__global__ __launch_bounds__(128) void particle_prop_kernel(
    const __bf16* __restrict__ pe, const __bf16* __restrict__ agg,
    const __bf16* __restrict__ pp_w, const float* __restrict__ pp_b,
    __bf16* __restrict__ effect) {
  int wave = threadIdx.x >> 5, lane = threadIdx.x & 31;
  int m = lane & 15, hi = (lane >> 4) & 1, n = m;
  long rowbase = ((long)blockIdx.x * 4 + wave) * 16;
  long row = rowbase + m;

  v8f acc[8];
#pragma unroll
  for (int nt = 0; nt < 8; ++nt) acc[nt] = vzero8();
  for (int kt = 0; kt < 8; ++kt) {
    int kb = kt * 32;
    const __bf16* arow = (kb < NF) ? (pe + row * NF + kb) : (agg + row * NF + (kb - NF));
    v16bf a = frag_ld(arow, 0, hi);
#pragma unroll
    for (int nt = 0; nt < 8; ++nt)
      acc[nt] = wmma_bf16(a, frag_ld(pp_w + (nt * 16 + n) * (2 * NF), kb, hi), acc[nt]);
  }
#pragma unroll
  for (int nt = 0; nt < 8; ++nt) {
    float bb = pp_b[nt * 16 + n];
#pragma unroll
    for (int r = 0; r < 8; ++r)
      effect[(rowbase + r + 8 * hi) * NF + nt * 16 + n] =
          (__bf16)fmaxf(acc[nt][r] + bb, 0.0f);
  }
}

// Rigid head: pooled mean of effect[:RIG] -> MLP 128->128->128->7 -> tvec
__global__ __launch_bounds__(128) void rigid_head_kernel(
    const __bf16* __restrict__ effect, const float* __restrict__ w0,
    const float* __restrict__ b0, const float* __restrict__ w1,
    const float* __restrict__ b1, const float* __restrict__ w2,
    const float* __restrict__ b2, float* __restrict__ tvec) {
  __shared__ float h0[NF], h1[NF];
  int t = threadIdx.x;
  float s = 0.0f;
  for (int r = 0; r < RIGC; ++r) s += (float)effect[(long)r * NF + t];
  h0[t] = s * (1.0f / RIGC);
  __syncthreads();
  float acc = b0[t];
  for (int k = 0; k < NF; ++k) acc += w0[t * NF + k] * h0[k];
  h1[t] = fmaxf(acc, 0.0f);
  __syncthreads();
  acc = b1[t];
  for (int k = 0; k < NF; ++k) acc += w1[t * NF + k] * h1[k];
  __syncthreads();
  h0[t] = fmaxf(acc, 0.0f);
  __syncthreads();
  if (t < 7) {
    float o = b2[t];
    for (int k = 0; k < NF; ++k) o += w2[t * NF + k] * h0[k];
    tvec[t] = o;
  }
}

// Rigid prediction: quaternion rotation about centroid (MEAN/STD are identity)
__global__ __launch_bounds__(256) void rigid_predict_kernel(
    const float* __restrict__ state, const float* __restrict__ tvec,
    const float* __restrict__ cent, float* __restrict__ out) {
  int p = blockIdx.x * 256 + threadIdx.x;
  if (p >= RIGC) return;
  float q0 = tvec[0], q1 = tvec[1], q2 = tvec[2], q3 = tvec[3];
  float inv = rsqrtf(q0 * q0 + q1 * q1 + q2 * q2 + q3 * q3);
  float w = q0 * inv, x = q1 * inv, y = q2 * inv, z = q3 * inv;
  float R00 = 1.0f - 2.0f * (y * y + z * z), R01 = 2.0f * (x * y + z * w), R02 = 2.0f * (x * z - y * w);
  float R10 = 2.0f * (x * y - z * w), R11 = 1.0f - 2.0f * (x * x + z * z), R12 = 2.0f * (y * z + x * w);
  float R20 = 2.0f * (x * z + y * w), R21 = 2.0f * (y * z - x * w), R22 = 1.0f - 2.0f * (x * x + y * y);
  float bx = tvec[4], by = tvec[5], bz = tvec[6];
  float cx = cent[0], cy = cent[1], cz = cent[2];
  float p0x = state[p * 6 + 0], p0y = state[p * 6 + 1], p0z = state[p * 6 + 2];
  float dx = p0x - cx, dy = p0y - cy, dz = p0z - cz;
  float p1x = dx * R00 + dy * R10 + dz * R20 + bx + cx;
  float p1y = dx * R01 + dy * R11 + dz * R21 + by + cy;
  float p1z = dx * R02 + dy * R12 + dz * R22 + bz + cz;
  out[p * 3 + 0] = (p1x - p0x) * DT_INV;
  out[p * 3 + 1] = (p1y - p0y) * DT_INV;
  out[p * 3 + 2] = (p1z - p0z) * DT_INV;
}

// Fluid head: 128->128->128->3 on effect[RIG:], last layer no relu
__global__ __launch_bounds__(128) void fluid_kernel(
    const __bf16* __restrict__ effect, const __bf16* __restrict__ w0,
    const float* __restrict__ b0, const __bf16* __restrict__ w1,
    const float* __restrict__ b1, const __bf16* __restrict__ w2,
    const float* __restrict__ b2, float* __restrict__ out) {
  __shared__ __bf16 lds[4][16 * NF];
  int wave = threadIdx.x >> 5, lane = threadIdx.x & 31;
  int m = lane & 15, hi = (lane >> 4) & 1, n = m;
  __bf16* t = lds[wave];
  long rowbase = RIGC + ((long)blockIdx.x * 4 + wave) * 16;
  long row = rowbase + m;

  // Layer 0 from global effect (bf16)
  v8f acc[8];
#pragma unroll
  for (int nt = 0; nt < 8; ++nt) acc[nt] = vzero8();
  for (int kt = 0; kt < 4; ++kt) {
    v16bf a = frag_ld(effect + row * NF, kt * 32, hi);
#pragma unroll
    for (int nt = 0; nt < 8; ++nt)
      acc[nt] = wmma_bf16(a, frag_ld(w0 + (nt * 16 + n) * NF, kt * 32, hi), acc[nt]);
  }
#pragma unroll
  for (int nt = 0; nt < 8; ++nt) {
    float bb = b0[nt * 16 + n];
#pragma unroll
    for (int r = 0; r < 8; ++r)
      t[(r + 8 * hi) * NF + nt * 16 + n] = (__bf16)fmaxf(acc[nt][r] + bb, 0.0f);
  }
  __syncthreads();

  // Layer 1 from LDS
#pragma unroll
  for (int nt = 0; nt < 8; ++nt) acc[nt] = vzero8();
  for (int kt = 0; kt < 4; ++kt) {
    v16bf af = frag_ld(t + m * NF, kt * 32, hi);
#pragma unroll
    for (int nt = 0; nt < 8; ++nt)
      acc[nt] = wmma_bf16(af, frag_ld(w1 + (nt * 16 + n) * NF, kt * 32, hi), acc[nt]);
  }
  __syncthreads();
#pragma unroll
  for (int nt = 0; nt < 8; ++nt) {
    float bb = b1[nt * 16 + n];
#pragma unroll
    for (int r = 0; r < 8; ++r)
      t[(r + 8 * hi) * NF + nt * 16 + n] = (__bf16)fmaxf(acc[nt][r] + bb, 0.0f);
  }
  __syncthreads();

  // Layer 2: Out = 3 (one n-tile, Outpad = 16, no relu)
  v8f c = vzero8();
  for (int kt = 0; kt < 4; ++kt)
    c = wmma_bf16(frag_ld(t + m * NF, kt * 32, hi), frag_ld(w2 + n * NF, kt * 32, hi), c);
  if (n < 3) {
    float bb = b2[n];
#pragma unroll
    for (int r = 0; r < 8; ++r) out[(rowbase + r + 8 * hi - RIGC) * 3 + n] = c[r] + bb;
  }
}

// ---------------------------------------------------------------------------
extern "C" void kernel_launch(void* const* d_in, const int* in_sizes, int n_in,
                              void* d_out, int out_size, void* d_ws, size_t ws_size,
                              hipStream_t stream) {
  (void)in_sizes; (void)n_in; (void)out_size; (void)ws_size;
  const float* state = (const float*)d_in[0];
  const float* attr  = (const float*)d_in[1];
  const float* Ra    = (const float*)d_in[2];
  const int*   recv  = (const int*)d_in[3];
  const int*   send  = (const int*)d_in[4];
  const float* pe_w0 = (const float*)d_in[5];
  const float* pe_b0 = (const float*)d_in[6];
  const float* pe_w1 = (const float*)d_in[7];
  const float* pe_b1 = (const float*)d_in[8];
  const float* re_w0 = (const float*)d_in[9];
  const float* re_b0 = (const float*)d_in[10];
  const float* re_w1 = (const float*)d_in[11];
  const float* re_b1 = (const float*)d_in[12];
  const float* re_w2 = (const float*)d_in[13];
  const float* re_b2 = (const float*)d_in[14];
  const float* rp_w  = (const float*)d_in[15];
  const float* rp_b  = (const float*)d_in[16];
  const float* pp_w  = (const float*)d_in[17];
  const float* pp_b  = (const float*)d_in[18];
  const float* rg_w0 = (const float*)d_in[19];
  const float* rg_b0 = (const float*)d_in[20];
  const float* rg_w1 = (const float*)d_in[21];
  const float* rg_b1 = (const float*)d_in[22];
  const float* rg_w2 = (const float*)d_in[23];
  const float* rg_b2 = (const float*)d_in[24];
  const float* fl_w0 = (const float*)d_in[25];
  const float* fl_b0 = (const float*)d_in[26];
  const float* fl_w1 = (const float*)d_in[27];
  const float* fl_b1 = (const float*)d_in[28];
  const float* fl_w2 = (const float*)d_in[29];
  const float* fl_b2 = (const float*)d_in[30];

  float* out = (float*)d_out;
  float* ws_f = (float*)d_ws;
  float* cent = ws_f;        // 8 floats
  float* tvec = ws_f + 8;    // 8 floats
  float* agg  = ws_f + 16;   // N*NF f32 (atomic scatter target)

  // bf16 region (16-byte aligned: (16 + N*NF)*4 is a multiple of 16)
  __bf16* bfp = (__bf16*)(ws_f + 16 + (size_t)NPART * NF);
  __bf16* pe_bf     = bfp; bfp += (size_t)NPART * NF;
  __bf16* effect_bf = bfp; bfp += (size_t)NPART * NF;
  __bf16* agg_bf    = bfp; bfp += (size_t)NPART * NF;
  __bf16* rel_bf    = bfp; bfp += (size_t)NEDGE * NF;
  __bf16* w_pe0 = bfp; bfp += 128 * 32;
  __bf16* w_pe1 = bfp; bfp += 128 * 128;
  __bf16* w_re0 = bfp; bfp += 128 * 32;
  __bf16* w_re1 = bfp; bfp += 128 * 128;
  __bf16* w_re2 = bfp; bfp += 128 * 128;
  __bf16* w_rp  = bfp; bfp += 128 * 384;
  __bf16* w_pp  = bfp; bfp += 128 * 256;
  __bf16* w_fl0 = bfp; bfp += 128 * 128;
  __bf16* w_fl1 = bfp; bfp += 128 * 128;
  __bf16* w_fl2 = bfp; bfp += 16 * 128;

  // One-time (per launch) weight conversion to padded bf16 — tiny.
  convert_w_kernel<<<16, 256, 0, stream>>>(pe_w0, w_pe0, 128, 15, 128, 32);
  convert_w_kernel<<<64, 256, 0, stream>>>(pe_w1, w_pe1, 128, 128, 128, 128);
  convert_w_kernel<<<16, 256, 0, stream>>>(re_w0, w_re0, 128, 31, 128, 32);
  convert_w_kernel<<<64, 256, 0, stream>>>(re_w1, w_re1, 128, 128, 128, 128);
  convert_w_kernel<<<64, 256, 0, stream>>>(re_w2, w_re2, 128, 128, 128, 128);
  convert_w_kernel<<<192, 256, 0, stream>>>(rp_w, w_rp, 128, 384, 128, 384);
  convert_w_kernel<<<128, 256, 0, stream>>>(pp_w, w_pp, 128, 256, 128, 256);
  convert_w_kernel<<<64, 256, 0, stream>>>(fl_w0, w_fl0, 128, 128, 128, 128);
  convert_w_kernel<<<64, 256, 0, stream>>>(fl_w1, w_fl1, 128, 128, 128, 128);
  convert_w_kernel<<<8, 256, 0, stream>>>(fl_w2, w_fl2, 3, 128, 16, 128);

  zero_bf16_kernel<<<512, 256, 0, stream>>>(effect_bf, (long)NPART * NF);
  centroid_kernel<<<1, 256, 0, stream>>>(state, cent);
  particle_encode_kernel<<<NPART / 64, 128, 0, stream>>>(state, attr, cent, w_pe0,
                                                         pe_b0, w_pe1, pe_b1, pe_bf);
  rel_encode_kernel<<<NEDGE / 64, 128, 0, stream>>>(attr, state, Ra, recv, send, cent,
                                                    w_re0, re_b0, w_re1, re_b1, w_re2,
                                                    re_b2, rel_bf);
  for (int step = 0; step < 2; ++step) {
    zero_f32_kernel<<<512, 256, 0, stream>>>(agg, (long)NPART * NF);
    rel_prop_kernel<<<NEDGE / 128, 128, 0, stream>>>(rel_bf, effect_bf, recv, send,
                                                     w_rp, rp_b, agg);
    cvt_f32_bf16_kernel<<<512, 256, 0, stream>>>(agg, agg_bf, (long)NPART * NF);
    particle_prop_kernel<<<NPART / 64, 128, 0, stream>>>(pe_bf, agg_bf, w_pp, pp_b,
                                                         effect_bf);
  }
  rigid_head_kernel<<<1, 128, 0, stream>>>(effect_bf, rg_w0, rg_b0, rg_w1, rg_b1,
                                           rg_w2, rg_b2, tvec);
  rigid_predict_kernel<<<RIGC / 256, 256, 0, stream>>>(state, tvec, cent, out);
  fluid_kernel<<<(NPART - RIGC) / 64, 128, 0, stream>>>(effect_bf, w_fl0, fl_b0, w_fl1,
                                                        fl_b1, w_fl2, fl_b2,
                                                        out + (size_t)RIGC * 3);
}
